// LSTMModel_67697274519892
// MI455X (gfx1250) — matrix operand order repaired and checked by reading, not compiled, not run on hardware
//
#include <hip/hip_runtime.h>

// ---------------------------------------------------------------------------
// Types for gfx1250 WMMA (wave32): v_wmma_f32_16x16x32_bf16
// ---------------------------------------------------------------------------
typedef __attribute__((ext_vector_type(16))) __bf16 bf16x16;
typedef __attribute__((ext_vector_type(8)))  __bf16 bf16x8;
typedef __attribute__((ext_vector_type(8)))  float  f32x8;

#define DEVI __device__ __forceinline__

DEVI __bf16 f2bf(float f) {
  unsigned u = __builtin_bit_cast(unsigned, f);
  unsigned r = u + 0x7FFFu + ((u >> 16) & 1u);   // round-to-nearest-even
  unsigned short h = (unsigned short)(r >> 16);
  return __builtin_bit_cast(__bf16, h);
}

DEVI float sigm(float x) { return 1.0f / (1.0f + __expf(-x)); }
DEVI float tanh_fast(float x) { return 2.0f * sigm(2.0f * x) - 1.0f; }

union AFrag { bf16x16 v; bf16x8 h[2]; };

// LDS byte offset of a generic pointer that points into __shared__ memory.
DEVI unsigned lds_off(const void* p) {
  return (unsigned)(__SIZE_TYPE__)(__attribute__((address_space(3))) const void*)p;
}

// gfx1250 async global -> LDS copy, 16 bytes per lane (ASYNCcnt-tracked).
DEVI void async_g2l_b128(unsigned lds_byte_addr, const void* gptr) {
  asm volatile("global_load_async_to_lds_b128 %0, %1, off"
               :: "v"(lds_byte_addr),
                  "v"((unsigned long long)(__SIZE_TYPE__)gptr)
               : "memory");
}
DEVI void wait_async0() { asm volatile("s_wait_asynccnt 0x0" ::: "memory"); }

DEVI f32x8 wmma_bf16(bf16x16 a, bf16x16 b, f32x8 c) {
  return __builtin_amdgcn_wmma_f32_16x16x32_bf16(false, a, false, b,
                                                 (short)0, c, false, false);
}

// ---------------------------------------------------------------------------
// Small helper kernels
// ---------------------------------------------------------------------------
__global__ void zero_u32_kernel(unsigned* __restrict__ p, int n) {
  int i = blockIdx.x * blockDim.x + threadIdx.x;
  if (i < n) p[i] = 0u;
}

__global__ void f32_to_bf16_kernel(const float* __restrict__ in,
                                   __bf16* __restrict__ out, int n) {
  int i = blockIdx.x * blockDim.x + threadIdx.x;
  if (i < n) out[i] = f2bf(in[i]);
}

__global__ void bias_sum_kernel(const float* __restrict__ a,
                                const float* __restrict__ b,
                                float* __restrict__ out, int n) {
  int i = blockIdx.x * blockDim.x + threadIdx.x;
  if (i < n) out[i] = a[i] + b[i];
}

// out[row, e] = bf16(emb[idx[row], e]);  rows = B*S, row-major [B,S,E]
__global__ void embed_gather_kernel(const int* __restrict__ idx,
                                    const float* __restrict__ emb,
                                    __bf16* __restrict__ out,
                                    int rows, int E_) {
  int i = blockIdx.x * blockDim.x + threadIdx.x;
  if (i >= rows * E_) return;
  int row = i / E_, e = i - row * E_;
  out[i] = f2bf(emb[(size_t)idx[row] * E_ + e]);
}

// ---------------------------------------------------------------------------
// GEMM: C[M,N] = A[M,K](bf16) @ W[N,K](bf16)^T + bias[N], C fp32 row-major.
// Block = 256 threads = 8 waves arranged 2(M) x 4(N); wave tile = 16x64.
// Grid = (M/32, N/256). K multiple of 64.
// A tile (32 contiguous rows x K) is staged to LDS with async-to-LDS copies;
// K-loop is unrolled x2 with all loads batched before the 8 WMMAs.
// ---------------------------------------------------------------------------
__global__ __launch_bounds__(256) void gemm_bf16_bias_kernel(
    const __bf16* __restrict__ A, const __bf16* __restrict__ W,
    const float* __restrict__ bias, float* __restrict__ C,
    int M, int N, int K) {
  __shared__ __bf16 Ash[32 * 512];   // 32 KB max (K <= 512)

  const int tid  = threadIdx.x;
  const int lane = tid & 31;
  const int wave = tid >> 5;
  const int wm   = wave & 1;         // M half of block tile
  const int wn   = wave >> 1;        // 0..3: N quarter
  const int half = lane >> 4;        // lane group
  const int r    = lane & 15;

  const int m0 = blockIdx.x * 32 + wm * 16;
  const int n0 = blockIdx.y * 256 + wn * 64;

  // ---- async stage of the block's A tile (contiguous 32*K elements) ----
  {
    const __bf16* Ablk = A + (size_t)blockIdx.x * 32 * K;
    const unsigned base = lds_off(Ash);
    const int chunks = (32 * K) / 8;            // 8 bf16 per b128
    for (int i = tid; i < chunks; i += 256)
      async_g2l_b128(base + (unsigned)i * 16u, Ablk + (size_t)i * 8);
    wait_async0();
  }
  __syncthreads();

  f32x8 acc[4] = {{}, {}, {}, {}};
  const __bf16* arow = Ash + (size_t)(wm * 16 + r) * K + half * 16;  // LDS
  const __bf16* brow = W + (size_t)(n0 + r) * K + half * 16;         // global
  const size_t  wstep = (size_t)16 * K;   // 16 weight rows per n-tile

  for (int k0 = 0; k0 < K; k0 += 64) {
    if (k0 + 64 < K) __builtin_prefetch(brow + k0 + 64, 0, 3);
    // batch ALL loads for two K-steps, then issue 8 WMMAs
    AFrag a0, a1;
    a0.h[0] = *(const bf16x8*)(arow + k0);
    a0.h[1] = *(const bf16x8*)(arow + k0 + 8);
    a1.h[0] = *(const bf16x8*)(arow + k0 + 32);
    a1.h[1] = *(const bf16x8*)(arow + k0 + 40);
    bf16x16 b0[4], b1[4];
#pragma unroll
    for (int nt = 0; nt < 4; ++nt) {
      b0[nt] = *(const bf16x16*)(brow + (size_t)nt * wstep + k0);
      b1[nt] = *(const bf16x16*)(brow + (size_t)nt * wstep + k0 + 32);
    }
#pragma unroll
    for (int nt = 0; nt < 4; ++nt) acc[nt] = wmma_bf16(a0.v, b0[nt], acc[nt]);
#pragma unroll
    for (int nt = 0; nt < 4; ++nt) acc[nt] = wmma_bf16(a1.v, b1[nt], acc[nt]);
  }

#pragma unroll
  for (int nt = 0; nt < 4; ++nt) {
    const int col = n0 + nt * 16 + r;
    const float bv = bias[col];
#pragma unroll
    for (int v = 0; v < 8; ++v) {
      const int row = m0 + v + 8 * half;
      C[(size_t)row * N + col] = acc[nt][v] + bv;
    }
  }
}

// ---------------------------------------------------------------------------
// Fused recurrent step:  g = Gx[t] + h_in @ Whh^T ;  gates ; update c, h.
// Grid = 8 blocks; block owns gate columns [j0, j0+64) of each gate chunk.
// h_in (contiguous 32 KB) is async-staged to LDS (shared by all 8 waves);
// the same LDS is reused (union) for the i/f/g/o gate exchange afterwards.
// ---------------------------------------------------------------------------
__global__ __launch_bounds__(256) void lstm_step_kernel(
    const __bf16* __restrict__ h_in,    // [32,512] bf16
    __bf16* __restrict__ h_out,         // [32,512] bf16
    float* __restrict__ c,              // [32,512] fp32, in-place
    const __bf16* __restrict__ Whh,     // [2048,512] bf16
    const float* __restrict__ Gx,       // pre-offset by t*2048
    int gx_bstride,                     // T*2048
    __bf16* __restrict__ seq,           // pre-offset by t*512
    int seq_bstride) {                  // T*512
  __shared__ union {
    __bf16 hsh[32 * 512];               // 32 KB staged h state
    float  gsh[4][32][65];              // ~33 KB gate exchange (+pad)
  } sh;

  const int tid  = threadIdx.x;
  const int lane = tid & 31;
  const int wave = tid >> 5;
  const int wm   = wave & 1;
  const int gate = wave >> 1;           // 0=i 1=f 2=g 3=o
  const int half = lane >> 4;
  const int r    = lane & 15;
  const int m0   = wm * 16;
  const int j0   = blockIdx.x * 64;
  const int nbase = gate * 512 + j0;

  // ---- async stage of h_in (32x512 bf16, contiguous) ----
  {
    const unsigned base = lds_off(sh.hsh);
    for (int i = tid; i < (32 * 512) / 8; i += 256)
      async_g2l_b128(base + (unsigned)i * 16u, h_in + (size_t)i * 8);
    wait_async0();
  }
  __syncthreads();

  f32x8 acc[4] = {{}, {}, {}, {}};
  const __bf16* arow = sh.hsh + (size_t)(m0 + r) * 512 + half * 16;  // LDS
  const __bf16* brow = Whh + (size_t)(nbase + r) * 512 + half * 16;  // global

  for (int k0 = 0; k0 < 512; k0 += 64) {
    AFrag a0, a1;
    a0.h[0] = *(const bf16x8*)(arow + k0);
    a0.h[1] = *(const bf16x8*)(arow + k0 + 8);
    a1.h[0] = *(const bf16x8*)(arow + k0 + 32);
    a1.h[1] = *(const bf16x8*)(arow + k0 + 40);
    bf16x16 b0[4], b1[4];
#pragma unroll
    for (int nt = 0; nt < 4; ++nt) {
      b0[nt] = *(const bf16x16*)(brow + (size_t)nt * 16 * 512 + k0);
      b1[nt] = *(const bf16x16*)(brow + (size_t)nt * 16 * 512 + k0 + 32);
    }
#pragma unroll
    for (int nt = 0; nt < 4; ++nt) acc[nt] = wmma_bf16(a0.v, b0[nt], acc[nt]);
#pragma unroll
    for (int nt = 0; nt < 4; ++nt) acc[nt] = wmma_bf16(a1.v, b1[nt], acc[nt]);
  }

  __syncthreads();   // everyone done reading sh.hsh before gsh overwrites it

#pragma unroll
  for (int nt = 0; nt < 4; ++nt) {
    const int j = nt * 16 + r;
#pragma unroll
    for (int v = 0; v < 8; ++v) {
      const int b = m0 + v + 8 * half;
      sh.gsh[gate][b][j] = acc[nt][v] + Gx[(size_t)b * gx_bstride + nbase + nt * 16 + r];
    }
  }
  __syncthreads();

  for (int i = tid; i < 32 * 64; i += 256) {
    const int b = i >> 6, j = i & 63;
    const float iv = sh.gsh[0][b][j];
    const float fv = sh.gsh[1][b][j];
    const float gv = sh.gsh[2][b][j];
    const float ov = sh.gsh[3][b][j];
    const size_t ci = (size_t)b * 512 + j0 + j;
    const float cn = sigm(fv) * c[ci] + sigm(iv) * tanh_fast(gv);
    const float hn = sigm(ov) * tanh_fast(cn);
    c[ci] = cn;
    const __bf16 hb = f2bf(hn);
    h_out[ci] = hb;
    seq[(size_t)b * seq_bstride + j0 + j] = hb;
  }
}

// ---------------------------------------------------------------------------
// Host orchestration
// ---------------------------------------------------------------------------
extern "C" void kernel_launch(void* const* d_in, const int* in_sizes, int n_in,
                              void* d_out, int out_size, void* d_ws, size_t ws_size,
                              hipStream_t stream) {
  (void)in_sizes; (void)n_in; (void)out_size; (void)ws_size;
  constexpr int B = 32, S = 128, T = 128, E = 256, H = 512, V = 32000;
  constexpr int G4 = 4 * H;  // 2048 gate columns

  const int*   src  = (const int*)d_in[0];
  const int*   tgt  = (const int*)d_in[1];
  const float* emb  = (const float*)d_in[2];
  const float* Wout = (const float*)d_in[3];
  const float* bout = (const float*)d_in[4];
  const float *Wih_f[4], *Whh_f[4], *bih_f[4], *bhh_f[4];  // enc0,enc1,dec0,dec1
  for (int s = 0; s < 4; ++s) {
    Wih_f[s] = (const float*)d_in[5 + 4 * s + 0];
    Whh_f[s] = (const float*)d_in[5 + 4 * s + 1];
    bih_f[s] = (const float*)d_in[5 + 4 * s + 2];
    bhh_f[s] = (const float*)d_in[5 + 4 * s + 3];
  }
  const int din[4] = {E, H, E, H};

  // ---- workspace carve (all 256B aligned; ~92 MB total) ----
  char* p = (char*)d_ws;
  auto carve = [&](size_t bytes) { char* q = p; p += (bytes + 255) & ~size_t(255); return q; };
  __bf16* Wout_bf = (__bf16*)carve((size_t)V * H * 2);
  __bf16* Wih_bf[4]; __bf16* Whh_bf[4]; float* bsum[4];
  for (int s = 0; s < 4; ++s) {
    Wih_bf[s] = (__bf16*)carve((size_t)G4 * din[s] * 2);
    Whh_bf[s] = (__bf16*)carve((size_t)G4 * H * 2);
    bsum[s]   = (float*)carve((size_t)G4 * 4);
  }
  __bf16* X    = (__bf16*)carve((size_t)B * S * E * 2);   // embedded inputs (reused)
  float*  Gx   = (float*)carve((size_t)B * T * G4 * 4);   // input projections (reused)
  __bf16* seqA = (__bf16*)carve((size_t)B * T * H * 2);   // layer-0 outputs
  __bf16* seqB = (__bf16*)carve((size_t)B * T * H * 2);   // layer-1 outputs
  __bf16* hbuf[2][2]; float* cbuf[2];
  for (int l = 0; l < 2; ++l) {
    hbuf[l][0] = (__bf16*)carve((size_t)B * H * 2);
    hbuf[l][1] = (__bf16*)carve((size_t)B * H * 2);
    cbuf[l]    = (float*)carve((size_t)B * H * 4);
  }

  auto gridFor = [](size_t n) { return dim3((unsigned)((n + 255) / 256)); };

  // ---- weight conversion + bias sums ----
  f32_to_bf16_kernel<<<gridFor((size_t)V * H), 256, 0, stream>>>(Wout, Wout_bf, V * H);
  for (int s = 0; s < 4; ++s) {
    f32_to_bf16_kernel<<<gridFor((size_t)G4 * din[s]), 256, 0, stream>>>(Wih_f[s], Wih_bf[s], G4 * din[s]);
    f32_to_bf16_kernel<<<gridFor((size_t)G4 * H), 256, 0, stream>>>(Whh_f[s], Whh_bf[s], G4 * H);
    bias_sum_kernel<<<gridFor(G4), 256, 0, stream>>>(bih_f[s], bhh_f[s], bsum[s], G4);
  }
  // ---- zero initial states (every call: deterministic) ----
  for (int l = 0; l < 2; ++l) {
    zero_u32_kernel<<<gridFor(B * H / 2), 256, 0, stream>>>((unsigned*)hbuf[l][0], B * H / 2);
    zero_u32_kernel<<<gridFor(B * H / 2), 256, 0, stream>>>((unsigned*)hbuf[l][1], B * H / 2);
    zero_u32_kernel<<<gridFor(B * H), 256, 0, stream>>>((unsigned*)cbuf[l], B * H);
  }

  int par[2] = {0, 0};  // h double-buffer parity per layer (persists enc->dec)
  auto run_pass = [&](int s, const __bf16* Ain, int Kin, __bf16* seqOut) {
    dim3 g((B * T) / 32, G4 / 256);
    gemm_bf16_bias_kernel<<<g, 256, 0, stream>>>(Ain, Wih_bf[s], bsum[s], Gx, B * T, G4, Kin);
    const int layer = s & 1;
    for (int t = 0; t < T; ++t) {
      lstm_step_kernel<<<8, 256, 0, stream>>>(
          hbuf[layer][par[layer]], hbuf[layer][par[layer] ^ 1], cbuf[layer],
          Whh_bf[s], Gx + (size_t)t * G4, T * G4,
          seqOut + (size_t)t * H, T * H);
      par[layer] ^= 1;
    }
  };

  // ---- encoder ----
  embed_gather_kernel<<<gridFor((size_t)B * S * E), 256, 0, stream>>>(src, emb, X, B * S, E);
  run_pass(0, X, E, seqA);      // enc layer 0
  run_pass(1, seqA, H, seqB);   // enc layer 1 (outputs unused; final h,c kept)
  // ---- decoder (states continue from encoder finals) ----
  embed_gather_kernel<<<gridFor((size_t)B * T * E), 256, 0, stream>>>(tgt, emb, X, B * T, E);
  run_pass(2, X, E, seqA);      // dec layer 0
  run_pass(3, seqA, H, seqB);   // dec layer 1 -> [B,T,H] bf16
  // ---- logits = seqB @ Wout^T + bout -> d_out [B,T,V] fp32 ----
  dim3 gp((B * T) / 32, V / 256);
  gemm_bf16_bias_kernel<<<gp, 256, 0, stream>>>(seqB, Wout_bf, bout, (float*)d_out, B * T, V, H);
}
